// AdaptiveMultiHeadAttention_76459007803728
// MI455X (gfx1250) — matrix operand order repaired and checked by reading, not compiled, not run on hardware
//
#include <hip/hip_runtime.h>
#include <cstddef>

// ---------------------------------------------------------------------------
// AdaptiveMultiHeadAttention on MI455X (gfx1250).
// f16 WMMA (16x16x32, fp32 accumulate) for all matmuls; f16 intermediates.
// Softmax cross-lane reductions via DPP16 ROW_XMASK butterflies (VALU-only).
// B=2, S=2048, E=1024, H=16, D=64.
// ---------------------------------------------------------------------------

typedef _Float16 f16;
typedef __attribute__((ext_vector_type(2)))  _Float16 v2h;
typedef __attribute__((ext_vector_type(4)))  _Float16 v4h;
typedef __attribute__((ext_vector_type(8)))  _Float16 v8h;
typedef __attribute__((ext_vector_type(16))) _Float16 v16h;
typedef __attribute__((ext_vector_type(8)))  float    v8f;

#define WMMA_F16(a, b, c) \
  __builtin_amdgcn_wmma_f32_16x16x32_f16(false, (a), false, (b), (short)0, (c), false, false)

static __device__ inline v16h cat8(v8h a, v8h b) {
  return __builtin_shufflevector(a, b, 0, 1, 2, 3, 4, 5, 6, 7, 8, 9, 10, 11, 12, 13, 14, 15);
}

// DPP16 ROW_XMASK lane-xor shuffle within 16-lane rows (no LDS round-trip).
template <int XMASK>
static __device__ inline float dpp_xor(float x) {
  return __int_as_float(__builtin_amdgcn_update_dpp(
      0, __float_as_int(x), 0x160 | XMASK, 0xf, 0xf, true));
}
// reduce over the 16-lane lo-group; result uniform across the group
static __device__ inline float rowmax16(float x) {
  x = fmaxf(x, dpp_xor<8>(x));
  x = fmaxf(x, dpp_xor<4>(x));
  x = fmaxf(x, dpp_xor<2>(x));
  x = fmaxf(x, dpp_xor<1>(x));
  return x;
}
static __device__ inline float rowsum16(float x) {
  x += dpp_xor<8>(x);
  x += dpp_xor<4>(x);
  x += dpp_xor<2>(x);
  x += dpp_xor<1>(x);
  return x;
}

static constexpr int Bc = 2, Sc = 2048, Ec = 1024, Hc = 16, Dc = 64;

// ---------------------------------------------------------------------------
// fp32 -> f16 conversion (x, qkv_w, out_w); n % 4 == 0
// ---------------------------------------------------------------------------
__global__ __launch_bounds__(256) void cvt_f32_to_f16(const float* __restrict__ in,
                                                      f16* __restrict__ out, int n) {
  int i = (blockIdx.x * 256 + threadIdx.x) * 4;
  if (i < n) {
    const float4 v = *(const float4*)(in + i);
    v4h h;
    h.x = (f16)v.x; h.y = (f16)v.y; h.z = (f16)v.z; h.w = (f16)v.w;
    *(v4h*)(out + i) = h;
  }
}

// ---------------------------------------------------------------------------
// head gating: mask[h] = sigmoid(g*imp) > thr ? 1 : 0
// ---------------------------------------------------------------------------
__global__ void head_mask_kernel(const float* __restrict__ g,
                                 const float* __restrict__ imp,
                                 const float* __restrict__ thr,
                                 float* __restrict__ mask) {
  int h = threadIdx.x;
  if (h < Hc) {
    float gs = 1.0f / (1.0f + __expf(-g[h] * imp[h]));
    mask[h] = (gs > thr[0]) ? 1.0f : 0.0f;
  }
}

// ---------------------------------------------------------------------------
// WMMA GEMM:  C[M,N] = A[M,K] @ W[N,K]^T + bias[N], optional per-head column
// mask. A/W are f16; accumulate fp32; output f16 (HALF_OUT) or fp32.
// Block tile 128x128, BK=32, 8 waves, each wave 32x64 (2x4 WMMA tiles).
// Software-pipelined: prefetch tile k+1 into VGPRs while computing tile k.
// Requires M%128==0, N%128==0, K%32==0.
// ---------------------------------------------------------------------------
template <bool HALF_OUT>
__global__ __launch_bounds__(256) void gemm_nt_f16(
    const f16* __restrict__ A, const f16* __restrict__ W,
    const float* __restrict__ bias, const float* __restrict__ headMask,
    void* __restrict__ Cv, int M, int N, int K) {
  constexpr int P = 40;  // LDS row pitch in halves (32 + 8) -> conflict-free frags
  __shared__ f16 As[128 * P];
  __shared__ f16 Bs[128 * P];

  const int tid  = threadIdx.x;
  const int lane = tid & 31;
  const int w    = tid >> 5;
  const int lo   = lane & 15;
  const int hi   = lane >> 4;
  const int m_off = (w & 3) * 32;   // 4 waves along M
  const int n_off = (w >> 2) * 64;  // 2 waves along N
  const int rowBase = blockIdx.x * 128;
  const int colBase = blockIdx.y * 128;

  // per-thread staging chunks: tile is 128x32 halves = 512 v8h chunks, 2/thread
  int cr[2], cc[2];
#pragma unroll
  for (int i = 0; i < 2; ++i) {
    int idx = tid + i * 256;
    cr[i] = idx >> 2;
    cc[i] = (idx & 3) << 3;
  }

  v8f acc[2][4] = {};
  v8h pa[2], pb[2];

  // prologue: fetch tile 0 into registers
#pragma unroll
  for (int i = 0; i < 2; ++i) {
    pa[i] = *(const v8h*)(A + (size_t)(rowBase + cr[i]) * K + cc[i]);
    pb[i] = *(const v8h*)(W + (size_t)(colBase + cr[i]) * K + cc[i]);
  }

  for (int kt = 0; kt < K; kt += 32) {
    // commit staged registers to LDS
#pragma unroll
    for (int i = 0; i < 2; ++i) {
      *(v8h*)&As[cr[i] * P + cc[i]] = pa[i];
      *(v8h*)&Bs[cr[i] * P + cc[i]] = pb[i];
    }
    __syncthreads();

    // prefetch next tile (latency hidden behind the WMMA block below)
    if (kt + 32 < K) {
#pragma unroll
      for (int i = 0; i < 2; ++i) {
        pa[i] = *(const v8h*)(A + (size_t)(rowBase + cr[i]) * K + kt + 32 + cc[i]);
        pb[i] = *(const v8h*)(W + (size_t)(colBase + cr[i]) * K + kt + 32 + cc[i]);
      }
    }

    // fragments: A lane holds row (m+lo), K chunks [8hi,8hi+8) and [16+8hi,+8);
    //            B lane holds col (n+lo), K chunks [8hi,8hi+8) and [16+8hi,+8)
    v16h af[2], bf[4];
#pragma unroll
    for (int i = 0; i < 2; ++i) {
      const f16* r = &As[(m_off + 16 * i + lo) * P];
      af[i] = cat8(*(const v8h*)(r + 8 * hi), *(const v8h*)(r + 16 + 8 * hi));
    }
#pragma unroll
    for (int j = 0; j < 4; ++j) {
      const f16* r = &Bs[(n_off + 16 * j + lo) * P];
      bf[j] = cat8(*(const v8h*)(r + 8 * hi), *(const v8h*)(r + 16 + 8 * hi));
    }
#pragma unroll
    for (int i = 0; i < 2; ++i) {
#pragma unroll
      for (int j = 0; j < 4; ++j) {
        acc[i][j] = WMMA_F16(af[i], bf[j], acc[i][j]);
      }
    }
    __syncthreads();
  }

  // epilogue: C-layout row = r + 8*hi, col = lo
#pragma unroll
  for (int i = 0; i < 2; ++i) {
#pragma unroll
    for (int j = 0; j < 4; ++j) {
      int col = colBase + n_off + 16 * j + lo;
      float bi = bias[col];
      float mk = headMask ? headMask[(col & (Ec - 1)) >> 6] : 1.0f;
#pragma unroll
      for (int r = 0; r < 8; ++r) {
        int row = rowBase + m_off + 16 * i + r + 8 * hi;
        float v = (acc[i][j][r] + bi) * mk;
        if (HALF_OUT)
          ((f16*)Cv)[(size_t)row * N + col] = (f16)v;
        else
          ((float*)Cv)[(size_t)row * N + col] = v;
      }
    }
  }
}

// ---------------------------------------------------------------------------
// v.mean over sequence (f16 qkv -> fp32 vmean[b,h,d])
// ---------------------------------------------------------------------------
__global__ __launch_bounds__(256) void vmean_kernel(const f16* __restrict__ qkv,
                                                    float* __restrict__ vmean) {
  __shared__ float red[256];
  int bh = blockIdx.x;
  int b = bh >> 4, h = bh & 15;
  int d = threadIdx.x & 63, chunk = threadIdx.x >> 6;
  float s = 0.f;
  for (int t = chunk; t < Sc; t += 4)
    s += (float)qkv[((size_t)(b * Sc + t) * 3 + 2) * Ec + h * Dc + d];
  red[threadIdx.x] = s;
  __syncthreads();
  if (threadIdx.x < 64) {
    float tot = red[threadIdx.x] + red[threadIdx.x + 64] +
                red[threadIdx.x + 128] + red[threadIdx.x + 192];
    vmean[(size_t)(b * Hc + h) * Dc + threadIdx.x] = tot * (1.0f / Sc);
  }
}

// ---------------------------------------------------------------------------
// Flash attention, f16 WMMA. One WG = 8 waves = 128 query rows of one (b,h).
// K staged row-major, V staged TRANSPOSED ([d][key]) so P@V B-fragments are
// contiguous. P goes through per-wave LDS (C-layout -> A-frag). Softmax
// row statistics use DPP16 ROW_XMASK butterflies (pure VALU, no LDS waits).
// ---------------------------------------------------------------------------
__global__ __launch_bounds__(256) void flash_attn_f16(
    const f16* __restrict__ qkv, const float* __restrict__ vmean,
    f16* __restrict__ out) {
  constexpr int P = 72;  // row pitch in halves (64 + 8)
  __shared__ f16 Qs[128 * P];     // [qrow][d]
  __shared__ f16 Ks[64 * P];      // [key][d]
  __shared__ f16 Vt[64 * P];      // [d][key]
  __shared__ f16 Ps[8][16 * P];   // per-wave [prow][key]

  const int tid  = threadIdx.x;
  const int lane = tid & 31;
  const int w    = tid >> 5;
  const int lo   = lane & 15;
  const int hi   = lane >> 4;
  const int bh = blockIdx.x;
  const int b = bh >> 4, h = bh & 15;
  const int s0 = blockIdx.y * 128;
  const int m0 = 16 * w;  // this wave's 16 query rows within the 128-row block

  // load Q block [128 x 64] halves: 1024 v8h chunks, 4/thread
#pragma unroll
  for (int i = 0; i < 4; ++i) {
    int idx = tid + i * 256;
    int r = idx >> 3, c = (idx & 7) << 3;
    *(v8h*)&Qs[r * P + c] =
        *(const v8h*)(qkv + ((size_t)(b * Sc + s0 + r) * 3 + 0) * Ec + h * Dc + c);
  }

  // V-transpose staging coords: thread owns keys [key0,key0+2) x d [d0,d0+8)
  const int key0 = (tid & 31) * 2;
  const int d0   = (tid >> 5) * 8;

  float m_run[8], l_run[8];
#pragma unroll
  for (int r = 0; r < 8; ++r) { m_run[r] = -1e30f; l_run[r] = 0.f; }
  v8f oacc[4] = {};

  for (int kb = 0; kb < Sc; kb += 64) {
    __syncthreads();  // previous iteration done reading Ks/Vt

    // K block [64 x 64], row-major: 512 chunks, 2/thread
#pragma unroll
    for (int i = 0; i < 2; ++i) {
      int idx = tid + i * 256;
      int r = idx >> 3, c = (idx & 7) << 3;
      *(v8h*)&Ks[r * P + c] =
          *(const v8h*)(qkv + ((size_t)(b * Sc + kb + r) * 3 + 1) * Ec + h * Dc + c);
    }
    // V block, transposed into Vt[d][key]
    {
      v8h vr[2];
#pragma unroll
      for (int q = 0; q < 2; ++q)
        vr[q] = *(const v8h*)(qkv +
                 ((size_t)(b * Sc + kb + key0 + q) * 3 + 2) * Ec + h * Dc + d0);
#pragma unroll
      for (int j = 0; j < 8; ++j) {
        v2h col;
        col.x = vr[0][j]; col.y = vr[1][j];
        *(v2h*)&Vt[(d0 + j) * P + key0] = col;
      }
    }
    __syncthreads();

    // scores = Q @ K^T (contraction over D=64, two K-steps of 32)
    v8f sacc[4] = {};
#pragma unroll
    for (int ks = 0; ks < 64; ks += 32) {
      const f16* qa = &Qs[(m0 + lo) * P + ks];
      v16h a = cat8(*(const v8h*)(qa + 8 * hi), *(const v8h*)(qa + 16 + 8 * hi));
#pragma unroll
      for (int t = 0; t < 4; ++t) {
        const f16* kp = &Ks[(16 * t + lo) * P + ks];
        v16h bfr = cat8(*(const v8h*)(kp + 8 * hi), *(const v8h*)(kp + 16 + 8 * hi));
        sacc[t] = WMMA_F16(a, bfr, sacc[t]);
      }
    }

    // online softmax; row = r + 8*hi, row elements live in the 16-lane lo group
#pragma unroll
    for (int r = 0; r < 8; ++r) {
      float cm = fmaxf(fmaxf(sacc[0][r], sacc[1][r]), fmaxf(sacc[2][r], sacc[3][r]));
      cm = rowmax16(cm);
      float nm = fmaxf(m_run[r], cm);
      float scale = __expf(m_run[r] - nm);
      float p = 0.f;
#pragma unroll
      for (int t = 0; t < 4; ++t) {
        float e = __expf(sacc[t][r] - nm);
        sacc[t][r] = e;
        p += e;
      }
      l_run[r] = l_run[r] * scale + rowsum16(p);
      m_run[r] = nm;
#pragma unroll
      for (int t = 0; t < 4; ++t) oacc[t][r] *= scale;
    }

    // stage P as f16 (per-wave, same-wave LDS ordering is in-order)
#pragma unroll
    for (int t = 0; t < 4; ++t) {
#pragma unroll
      for (int r = 0; r < 8; ++r) {
        Ps[w][(r + 8 * hi) * P + 16 * t + lo] = (f16)sacc[t][r];
      }
    }

    // out += P @ V (contraction over 64 keys, two K-steps of 32)
#pragma unroll
    for (int ks = 0; ks < 64; ks += 32) {
      const f16* pp = &Ps[w][lo * P + ks];
      v16h a = cat8(*(const v8h*)(pp + 8 * hi), *(const v8h*)(pp + 16 + 8 * hi));
#pragma unroll
      for (int t = 0; t < 4; ++t) {
        const f16* vb = &Vt[(16 * t + lo) * P + ks];
        v16h bfr = cat8(*(const v8h*)(vb + 8 * hi), *(const v8h*)(vb + 16 + 8 * hi));
        oacc[t] = WMMA_F16(a, bfr, oacc[t]);
      }
    }
  }

  // finalize: divide by softmax denom, add vmean residual, store f16
  const float* vm = vmean + (size_t)(b * Hc + h) * Dc;
#pragma unroll
  for (int r = 0; r < 8; ++r) {
    float inv = 1.0f / l_run[r];
    int s = s0 + m0 + r + 8 * hi;
#pragma unroll
    for (int t = 0; t < 4; ++t) {
      int d = 16 * t + lo;
      out[(size_t)(b * Sc + s) * Ec + h * Dc + d] = (f16)(oacc[t][r] * inv + vm[d]);
    }
  }
}

// ---------------------------------------------------------------------------
// launch
// ---------------------------------------------------------------------------
extern "C" void kernel_launch(void* const* d_in, const int* in_sizes, int n_in,
                              void* d_out, int out_size, void* d_ws, size_t ws_size,
                              hipStream_t stream) {
  const float* x      = (const float*)d_in[0];  // [B,S,E]
  const float* qkv_w  = (const float*)d_in[1];  // [3E,E]
  const float* qkv_b  = (const float*)d_in[2];  // [3E]
  const float* out_w  = (const float*)d_in[3];  // [E,E]
  const float* out_b  = (const float*)d_in[4];  // [E]
  const float* gates  = (const float*)d_in[5];  // [H]
  const float* imp    = (const float*)d_in[6];  // [H]
  const float* thr    = (const float*)d_in[7];  // scalar
  (void)in_sizes; (void)n_in; (void)out_size; (void)ws_size;

  const int M   = Bc * Sc;          // 4096
  const int nX  = M * Ec;           // 4 Mi
  const int nQW = 3 * Ec * Ec;      // 3 Mi
  const int nOW = Ec * Ec;          // 1 Mi

  // workspace layout (bytes)
  char* ws = (char*)d_ws;
  float* mask  = (float*)ws;                               // 16 f32
  float* vmean = (float*)(ws + 4096);                      // 2048 f32
  size_t off = 65536;
  f16* xh    = (f16*)(ws + off); off += (size_t)nX * 2;    // 8 MB
  f16* qwh   = (f16*)(ws + off); off += (size_t)nQW * 2;   // 6 MB
  f16* owh   = (f16*)(ws + off); off += (size_t)nOW * 2;   // 2 MB
  f16* qkvh  = (f16*)(ws + off); off += (size_t)M * 3 * Ec * 2;  // 24 MB
  f16* attnh = (f16*)(ws + off);                           // 8 MB

  head_mask_kernel<<<1, 32, 0, stream>>>(gates, imp, thr, mask);
  cvt_f32_to_f16<<<nX / 1024, 256, 0, stream>>>(x, xh, nX);
  cvt_f32_to_f16<<<nQW / 1024, 256, 0, stream>>>(qkv_w, qwh, nQW);
  cvt_f32_to_f16<<<nOW / 1024, 256, 0, stream>>>(out_w, owh, nOW);

  // qkv = x @ qkv_w^T + b, masked per head -> f16
  {
    dim3 grid(M / 128, (3 * Ec) / 128);
    gemm_nt_f16<true><<<grid, 256, 0, stream>>>(xh, qwh, qkv_b, mask, qkvh,
                                                M, 3 * Ec, Ec);
  }

  vmean_kernel<<<dim3(Bc * Hc), 256, 0, stream>>>(qkvh, vmean);

  flash_attn_f16<<<dim3(Bc * Hc, Sc / 128), 256, 0, stream>>>(qkvh, vmean, attnh);

  // out = attn @ out_w^T + out_b -> fp32
  {
    dim3 grid(M / 128, Ec / 128);
    gemm_nt_f16<false><<<grid, 256, 0, stream>>>(attnh, owh, out_b, nullptr,
                                                 d_out, M, Ec, Ec);
  }
}
